// SelfAttention_85624468013281
// MI455X (gfx1250) — matrix operand order repaired and checked by reading, not compiled, not run on hardware
//
#include <hip/hip_runtime.h>
#include <hip/hip_bf16.h>
#include <cstdint>
#include <cstddef>

// ---------------------------------------------------------------------------
// Problem constants (match reference: B=4, T=2048, C=768, H=12, D=64)
// ---------------------------------------------------------------------------
#define B_DIM 4
#define T_DIM 2048
#define C_DIM 768
#define N_HEAD 12
#define D_HEAD 64
#define F_DIM (3 * C_DIM)        // 2304
#define M_DIM (B_DIM * T_DIM)    // 8192

typedef __attribute__((ext_vector_type(8)))  _Float16 v8h;
typedef __attribute__((ext_vector_type(16))) _Float16 v16h;
typedef __attribute__((ext_vector_type(8)))  float    v8f;

// 1/sqrt(D) softmax scale folded with log2(e) so we can use exp2 (v_exp_f32).
static constexpr float QK_SCALE = 0.125f * 1.44269504088896340736f;

// ---------------------------------------------------------------------------
// WMMA fragment layouts (CDNA5 ISA 7.12.2, wave32):
//
// A-frag (16x32 f16, M x K): lane m = lane&15 holds row m. With lh = lane>>4,
// element e holds K = e + 16*lh (this code's order, not the ISA table's)  ->  one contiguous 16-half (32-byte) chunk.
//
// B-frag (32x16 f16, K x N): lane holds column n = lane&15; element e holds
// K = e + 16*lh (this code's order, same as A) ->  one contiguous 16-half (32-byte) chunk.
//
// C/D-frag (16x16 f32): lane holds column n = lane&15; vgpr j holds row
// m = j + 8*lh.
// ---------------------------------------------------------------------------
__device__ __forceinline__ v16h load_a_frag(const _Float16* __restrict__ row, int lh) {
    return *reinterpret_cast<const v16h*>(row + 16 * lh);
}

__device__ __forceinline__ v8f wmma_f16(v16h a, v16h b, v8f c) {
    return __builtin_amdgcn_wmma_f32_16x16x32_f16(
        /*neg_a=*/false, a, /*neg_b=*/false, b,
        /*c_mod=*/(short)0, c, /*reuse_a=*/false, /*reuse_b=*/false);
}

// ---------------------------------------------------------------------------
// Kernel 0: fp32 -> f16 conversion
// ---------------------------------------------------------------------------
__global__ void cvt_f32_to_f16(const float* __restrict__ src,
                               _Float16* __restrict__ dst, int n) {
    int i = blockIdx.x * blockDim.x + threadIdx.x;
    if (i < n) dst[i] = (_Float16)src[i];
}

// ---------------------------------------------------------------------------
// Kernel 1: QKV projection.  qkv[t,f] = sum_c x[t,c] * w_attn[f,c]
// Each wave: 32 rows (t) x 64 cols (f) -> 8 WMMAs per 12 b128 loads per
// k-step.  4 waves/WG cover a 128x64 block.
// Epilogue routes to Q (scaled, [B,H,T,D]), K ([B,H,T,D]), V^T ([B,H,D,T]).
// ---------------------------------------------------------------------------
__global__ __launch_bounds__(128) void qkv_gemm(
    const _Float16* __restrict__ xh, const _Float16* __restrict__ wah,
    _Float16* __restrict__ qh, _Float16* __restrict__ kh,
    _Float16* __restrict__ vth) {
    const int lane = threadIdx.x & 31;
    const int wave = threadIdx.x >> 5;
    const int ml = lane & 15;
    const int lh = lane >> 4;
    const int tbase = blockIdx.x * 128 + wave * 32;
    const int nbase = blockIdx.y * 64;

    v8f acc[2][4] = {};
    const _Float16* arow0 = xh + (size_t)(tbase + ml) * C_DIM;
    const _Float16* arow1 = xh + (size_t)(tbase + 16 + ml) * C_DIM;

    for (int kb = 0; kb < C_DIM; kb += 32) {
        v16h a0 = load_a_frag(arow0 + kb, lh);
        v16h a1 = load_a_frag(arow1 + kb, lh);
#pragma unroll
        for (int nt = 0; nt < 4; ++nt) {
            const _Float16* brow =
                wah + (size_t)(nbase + nt * 16 + ml) * C_DIM + kb + 16 * lh;
            v16h b = *reinterpret_cast<const v16h*>(brow);
            acc[0][nt] = wmma_f16(a0, b, acc[0][nt]);
            acc[1][nt] = wmma_f16(a1, b, acc[1][nt]);
        }
    }

    const int b = tbase >> 11;                 // / T_DIM (tile within one b)
#pragma unroll
    for (int mt = 0; mt < 2; ++mt) {
        const int tloc0 = (tbase & (T_DIM - 1)) + mt * 16 + 8 * lh;
#pragma unroll
        for (int nt = 0; nt < 4; ++nt) {
            const int f = nbase + nt * 16 + ml;    // lane's output column
            const int which = f / C_DIM;           // uniform within tile
            const int fl = f - which * C_DIM;
            const int head = fl >> 6;
            const int d = fl & 63;
            const size_t bh = (size_t)(b * N_HEAD + head);
            if (which == 2) {
                // V^T [B,H,D,T]: fixed d, 8 consecutive t -> packed store
                _Float16* dst = vth + (bh * D_HEAD + d) * T_DIM + tloc0;
                v8h pk;
#pragma unroll
                for (int j = 0; j < 8; ++j) pk[j] = (_Float16)acc[mt][nt][j];
                *reinterpret_cast<v8h*>(dst) = pk;
            } else {
                _Float16* dst =
                    (which == 0 ? qh : kh) + (bh * T_DIM + tloc0) * D_HEAD + d;
                const float s = (which == 0) ? QK_SCALE : 1.0f;
#pragma unroll
                for (int j = 0; j < 8; ++j)
                    dst[(size_t)j * D_HEAD] = (_Float16)(acc[mt][nt][j] * s);
            }
        }
    }
}

// ---------------------------------------------------------------------------
// Kernel 2: flash attention (non-causal, matches reference).
// One wave = 16 queries, full D=64, 64 keys per iteration (16 WMMAs/iter).
// Computes S^T = K@Q^T and O^T = V^T@P^T so softmax is per-lane-column:
// one shfl_xor(16) reduction, scalar alpha, register-only P^T layout
// conversion, packed contiguous y stores.
// ---------------------------------------------------------------------------
__global__ __launch_bounds__(128) void attn_kernel(
    const _Float16* __restrict__ qh, const _Float16* __restrict__ kh,
    const _Float16* __restrict__ vth, _Float16* __restrict__ yh) {
    const int lane = threadIdx.x & 31;
    const int wave = threadIdx.x >> 5;
    const int q = lane & 15;       // query column (also A-frag row index)
    const int lh = lane >> 4;
    const int bh = blockIdx.x;     // b*H + h
    const int b = bh / N_HEAD;
    const int h = bh - b * N_HEAD;
    const int q0 = blockIdx.y * 64 + wave * 16;

    const _Float16* Qp = qh + (size_t)bh * T_DIM * D_HEAD;
    const _Float16* Kp = kh + (size_t)bh * T_DIM * D_HEAD;
    const _Float16* Vp = vth + (size_t)bh * D_HEAD * T_DIM;

    // Q^T B-frags (d 0..31 and 32..63), loaded once
    const _Float16* qrow = Qp + (size_t)(q0 + q) * D_HEAD;
    const v16h Qb0 = *reinterpret_cast<const v16h*>(qrow + 16 * lh);
    const v16h Qb1 = *reinterpret_cast<const v16h*>(qrow + 32 + 16 * lh);

    v8f O[4] = {};                  // O^T: 4 tiles of (16 d) x (16 q)
    float m_run = -__builtin_inff();
    float l_run = 0.0f;

    for (int k0 = 0; k0 < T_DIM; k0 += 64) {
        // --- S^T = K @ Q^T for 64 keys (four 16-key subtiles) ---
        v8f S[4];
#pragma unroll
        for (int s = 0; s < 4; ++s) {
            const _Float16* krow = Kp + (size_t)(k0 + s * 16 + q) * D_HEAD;
            v8f t = {};
            t = wmma_f16(load_a_frag(krow, lh), Qb0, t);
            t = wmma_f16(load_a_frag(krow + 32, lh), Qb1, t);
            S[s] = t;
        }

        // --- per-query (per-lane) running softmax over 64 keys ---
        float mp = S[0][0];
#pragma unroll
        for (int s = 0; s < 4; ++s)
#pragma unroll
            for (int j = 0; j < 8; ++j) mp = fmaxf(mp, S[s][j]);
        mp = fmaxf(mp, __shfl_xor(mp, 16, 32));
        const float mnew = fmaxf(m_run, mp);
        const float alpha = __builtin_exp2f(m_run - mnew);
        m_run = mnew;

        float ls = 0.0f;
#pragma unroll
        for (int s = 0; s < 4; ++s)
#pragma unroll
            for (int j = 0; j < 8; ++j) {
                S[s][j] = __builtin_exp2f(S[s][j] - mnew);
                ls += S[s][j];
            }
        ls += __shfl_xor(ls, 16, 32);
        l_run = l_run * alpha + ls;
#pragma unroll
        for (int dt = 0; dt < 4; ++dt)
#pragma unroll
            for (int j = 0; j < 8; ++j) O[dt][j] *= alpha;

        // --- P^T B-frags (keys 0..31 and 32..63 of this block) ---
        // lane(lh,q) element e needs P_{lh}[e&7] from lane (q + 16*(e>=8)).
        v16h Pb[2];
#pragma unroll
        for (int p = 0; p < 2; ++p) {
            const v8f P0 = S[2 * p];
            const v8f P1 = S[2 * p + 1];
#pragma unroll
            for (int j = 0; j < 8; ++j) {
                const float x0 = __shfl_xor(P0[j], 16, 32);
                const float x1 = __shfl_xor(P1[j], 16, 32);
                Pb[p][j]     = (_Float16)(lh ? x1 : P0[j]);
                Pb[p][j + 8] = (_Float16)(lh ? P1[j] : x0);
            }
        }

        // --- O^T += V^T @ P^T ---
#pragma unroll
        for (int dt = 0; dt < 4; ++dt) {
            const _Float16* vrow = Vp + (size_t)(dt * 16 + q) * T_DIM + k0;
            O[dt] = wmma_f16(load_a_frag(vrow, lh), Pb[0], O[dt]);
            O[dt] = wmma_f16(load_a_frag(vrow + 32, lh), Pb[1], O[dt]);
        }
    }

    // --- epilogue: y[b, t, h*64 + d] = O^T / l ; 8 consecutive d per lane ---
    const float invl = 1.0f / l_run;
    const size_t trow = (size_t)(b * T_DIM + q0 + q) * C_DIM + h * D_HEAD;
#pragma unroll
    for (int dt = 0; dt < 4; ++dt) {
        v8h pk;
#pragma unroll
        for (int j = 0; j < 8; ++j) pk[j] = (_Float16)(O[dt][j] * invl);
        *reinterpret_cast<v8h*>(yh + trow + dt * 16 + 8 * lh) = pk;
    }
}

// ---------------------------------------------------------------------------
// Kernel 3: output projection. out[t,o] = sum_c y[t,c] * w_proj[o,c], fp32 out
// Same 32x64 wave tile as the QKV GEMM.
// ---------------------------------------------------------------------------
__global__ __launch_bounds__(128) void proj_gemm(
    const _Float16* __restrict__ yh, const _Float16* __restrict__ wph,
    float* __restrict__ out) {
    const int lane = threadIdx.x & 31;
    const int wave = threadIdx.x >> 5;
    const int ml = lane & 15;
    const int lh = lane >> 4;
    const int tbase = blockIdx.x * 128 + wave * 32;
    const int nbase = blockIdx.y * 64;

    v8f acc[2][4] = {};
    const _Float16* arow0 = yh + (size_t)(tbase + ml) * C_DIM;
    const _Float16* arow1 = yh + (size_t)(tbase + 16 + ml) * C_DIM;

    for (int kb = 0; kb < C_DIM; kb += 32) {
        v16h a0 = load_a_frag(arow0 + kb, lh);
        v16h a1 = load_a_frag(arow1 + kb, lh);
#pragma unroll
        for (int nt = 0; nt < 4; ++nt) {
            const _Float16* brow =
                wph + (size_t)(nbase + nt * 16 + ml) * C_DIM + kb + 16 * lh;
            v16h bfr = *reinterpret_cast<const v16h*>(brow);
            acc[0][nt] = wmma_f16(a0, bfr, acc[0][nt]);
            acc[1][nt] = wmma_f16(a1, bfr, acc[1][nt]);
        }
    }

#pragma unroll
    for (int mt = 0; mt < 2; ++mt) {
#pragma unroll
        for (int nt = 0; nt < 4; ++nt) {
            const int o = nbase + nt * 16 + ml;
            float* dst =
                out + (size_t)(tbase + mt * 16 + 8 * lh) * C_DIM + o;
#pragma unroll
            for (int j = 0; j < 8; ++j) dst[(size_t)j * C_DIM] = acc[mt][nt][j];
        }
    }
}

// ---------------------------------------------------------------------------
// Host launcher
// ---------------------------------------------------------------------------
extern "C" void kernel_launch(void* const* d_in, const int* in_sizes, int n_in,
                              void* d_out, int out_size, void* d_ws,
                              size_t ws_size, hipStream_t stream) {
    const float* x = (const float*)d_in[0];       // [B,T,C]
    const float* w_attn = (const float*)d_in[1];  // [3C,C]
    const float* w_proj = (const float*)d_in[2];  // [C,C]
    float* out = (float*)d_out;                   // [B,T,C] fp32

    char* ws = (char*)d_ws;
    size_t off = 0;
    auto wsalloc = [&](size_t bytes) -> void* {
        void* p = ws + off;
        off += (bytes + 255) & ~(size_t)255;
        return p;
    };
    _Float16* xh  = (_Float16*)wsalloc((size_t)M_DIM * C_DIM * 2);
    _Float16* wah = (_Float16*)wsalloc((size_t)F_DIM * C_DIM * 2);
    _Float16* wph = (_Float16*)wsalloc((size_t)C_DIM * C_DIM * 2);
    _Float16* qh  = (_Float16*)wsalloc((size_t)M_DIM * C_DIM * 2);  // [B,H,T,D]
    _Float16* kh  = (_Float16*)wsalloc((size_t)M_DIM * C_DIM * 2);  // [B,H,T,D]
    _Float16* vth = (_Float16*)wsalloc((size_t)M_DIM * C_DIM * 2);  // [B,H,D,T]
    _Float16* yh  = (_Float16*)wsalloc((size_t)M_DIM * C_DIM * 2);  // [B,T,C]

    const int nx = M_DIM * C_DIM;
    const int na = F_DIM * C_DIM;
    const int np = C_DIM * C_DIM;
    cvt_f32_to_f16<<<(nx + 255) / 256, 256, 0, stream>>>(x, xh, nx);
    cvt_f32_to_f16<<<(na + 255) / 256, 256, 0, stream>>>(w_attn, wah, na);
    cvt_f32_to_f16<<<(np + 255) / 256, 256, 0, stream>>>(w_proj, wph, np);

    qkv_gemm<<<dim3(M_DIM / 128, F_DIM / 64), 128, 0, stream>>>(xh, wah, qh,
                                                                kh, vth);
    attn_kernel<<<dim3(B_DIM * N_HEAD, T_DIM / 64), 128, 0, stream>>>(qh, kh,
                                                                      vth, yh);
    proj_gemm<<<dim3(M_DIM / 128, C_DIM / 64), 128, 0, stream>>>(yh, wph, out);
}
